// HadamardTernaryLinear_54073638256918
// MI455X (gfx1250) — matrix operand order repaired and checked by reading, not compiled, not run on hardware
//
#include <hip/hip_runtime.h>

typedef __attribute__((ext_vector_type(16))) _Float16 v16h;
typedef __attribute__((ext_vector_type(8)))  _Float16 v8h;
typedef __attribute__((ext_vector_type(8)))  float    v8f;

#define G_DIM   32
#define IN_O    128
#define OUT_O   128
#define D_DIM   (G_DIM * IN_O)   // 4096
#define MTILE   16
#define LROW    (IN_O + 8)       // 136 halfs: 272B row stride breaks bank conflicts

__device__ __forceinline__ void fht32(float v[G_DIM]) {
#pragma unroll
  for (int s = 1; s < G_DIM; s <<= 1) {
#pragma unroll
    for (int g = 0; g < G_DIM; ++g) {
      if ((g & s) == 0) {
        float a = v[g], b = v[g | s];
        v[g] = a + b;
        v[g | s] = a - b;
      }
    }
  }
}

// ---------- weight absmean reduction ----------
__global__ void htl_abs_partial(const float* __restrict__ w, float* __restrict__ part, int n) {
  __shared__ float red[256];
  int tid = threadIdx.x;
  float s = 0.f;
  for (int i = blockIdx.x * 256 + tid; i < n; i += gridDim.x * 256) s += fabsf(w[i]);
  red[tid] = s;
  __syncthreads();
#pragma unroll
  for (int off = 128; off > 0; off >>= 1) {
    if (tid < off) red[tid] += red[tid + off];
    __syncthreads();
  }
  if (tid == 0) part[blockIdx.x] = red[0];
}

__global__ void htl_abs_final(const float* __restrict__ part, float* __restrict__ scalep,
                              float inv_n) {
  __shared__ float red[256];
  int tid = threadIdx.x;
  red[tid] = part[tid];
  __syncthreads();
#pragma unroll
  for (int off = 128; off > 0; off >>= 1) {
    if (tid < off) red[tid] += red[tid + off];
    __syncthreads();
  }
  if (tid == 0) scalep[0] = red[0] * inv_n + 1e-8f;
}

// ---------- ternary quantize to f16 {-1,0,1} ----------
__global__ void htl_quantize(const float* __restrict__ w, const float* __restrict__ scalep,
                             _Float16* __restrict__ wq, int n) {
  int i = blockIdx.x * 256 + threadIdx.x;
  if (i < n) {
    float q = rintf(w[i] / scalep[0]);          // round-half-even like jnp.round
    q = fminf(1.f, fmaxf(-1.f, q));
    wq[i] = (_Float16)q;
  }
}

// ---------- fused: alpha * FHT_g -> grouped WMMA -> FHT_h * beta*scale ----------
__launch_bounds__(256, 1)
__global__ void htl_fused(const float* __restrict__ x, const _Float16* __restrict__ wq,
                          const float* __restrict__ alpha, const float* __restrict__ beta,
                          const float* __restrict__ scalep, float* __restrict__ out) {
  __shared__ __align__(16) _Float16 xm[G_DIM][MTILE][LROW];  // 139264 B
  __shared__ __align__(16) _Float16 yp[G_DIM][MTILE][LROW];  // 139264 B

  const int  tid = threadIdx.x;
  const long tokBase = (long)blockIdx.x * MTILE;
  const float rs32 = 0.17677669529663687f;   // 1/sqrt(32), folded per FHT

  // ---- Phase 1: load x, *alpha, FHT over g, f16 -> LDS ----
  {
    const int i = tid & 127;
    const int th = tid >> 7;                  // 2 tokens per pass
    float av[G_DIM];
#pragma unroll
    for (int g = 0; g < G_DIM; ++g) av[g] = alpha[g * IN_O + i];
#pragma unroll 1
    for (int pass = 0; pass < 8; ++pass) {
      const int t = th + 2 * pass;
      const float* xr = x + (tokBase + t) * D_DIM;
      float v[G_DIM];
#pragma unroll
      for (int g = 0; g < G_DIM; ++g) v[g] = xr[g * IN_O + i] * av[g];
      fht32(v);
#pragma unroll
      for (int g = 0; g < G_DIM; ++g) xm[g][t][i] = (_Float16)(v[g] * rs32);
    }
  }
  __syncthreads();

  // ---- Phase 2: per-group 16x128 = (16x128)x(128x128) via WMMA f16 ----
  {
    const int lane = tid & 31;
    const int wid  = tid >> 5;                // 8 waves -> 4 groups each
    const int l16  = lane & 15;               // M row (A) / N col (B)
    const int sel  = lane >> 4;               // K-half selector
#pragma unroll 1
    for (int q = 0; q < 4; ++q) {
      const int h = wid + 8 * q;
      v8f acc[8];
#pragma unroll
      for (int nt = 0; nt < 8; ++nt) acc[nt] = (v8f){0.f,0.f,0.f,0.f,0.f,0.f,0.f,0.f};
#pragma unroll 1
      for (int kc = 0; kc < 4; ++kc) {
        // A fragment (16x32 f16): lanes 0-15 hold K {0..7,16..23}, lanes 16-31 {8..15,24..31}
        const _Float16* ap = &xm[h][l16][kc * 32 + sel * 8];
        v8h alo = *(const v8h*)ap;
        v8h ahi = *(const v8h*)(ap + 16);
        v16h afrag = __builtin_shufflevector(alo, ahi, 0,1,2,3,4,5,6,7,8,9,10,11,12,13,14,15);
#pragma unroll
        for (int nt = 0; nt < 8; ++nt) {
          // B fragment (32x16 f16): lane n, lanes 0-15 K {0..7,16..23}, lanes 16-31 {8..15,24..31}
          const _Float16* bp = wq + (((h * OUT_O) + nt * 16 + l16) * IN_O + kc * 32 + sel * 8);
          v8h blo = *(const v8h*)bp;
          v8h bhi = *(const v8h*)(bp + 16);
          v16h bfrag = __builtin_shufflevector(blo, bhi, 0,1,2,3,4,5,6,7,8,9,10,11,12,13,14,15);
          acc[nt] = __builtin_amdgcn_wmma_f32_16x16x32_f16(
              false, afrag, false, bfrag, (short)0, acc[nt], false, false);
        }
      }
      // D layout: VGPR r -> M = r + 8*sel, N = l16
#pragma unroll
      for (int nt = 0; nt < 8; ++nt) {
#pragma unroll
        for (int r = 0; r < 8; ++r) {
          yp[h][r + 8 * sel][nt * 16 + l16] = (_Float16)acc[nt][r];
        }
      }
    }
  }
  __syncthreads();

  // ---- Phase 3: FHT over h, * beta * wscale, store ----
  {
    const int o = tid & 127;
    const int th = tid >> 7;
    const float sc = scalep[0] * rs32;
    float bv[G_DIM];
#pragma unroll
    for (int g = 0; g < G_DIM; ++g) bv[g] = beta[g * OUT_O + o] * sc;
#pragma unroll 1
    for (int pass = 0; pass < 8; ++pass) {
      const int t = th + 2 * pass;
      float u[G_DIM];
#pragma unroll
      for (int hh = 0; hh < G_DIM; ++hh) u[hh] = (float)yp[hh][t][o];
      fht32(u);
      float* yr = out + (tokBase + t) * D_DIM;
#pragma unroll
      for (int g = 0; g < G_DIM; ++g) yr[g * OUT_O + o] = u[g] * bv[g];
    }
  }
}

extern "C" void kernel_launch(void* const* d_in, const int* in_sizes, int n_in,
                              void* d_out, int out_size, void* d_ws, size_t ws_size,
                              hipStream_t stream) {
  const float* x     = (const float*)d_in[0];
  const float* w     = (const float*)d_in[1];
  const float* alpha = (const float*)d_in[2];
  const float* beta  = (const float*)d_in[3];
  float*       out   = (float*)d_out;

  float*     part   = (float*)d_ws;                          // 256 floats
  float*     scalep = part + 256;                            // 1 float
  _Float16*  wq     = (_Float16*)((char*)d_ws + 4096);       // 1 MB ternary f16

  const int nW   = in_sizes[1];                // 32*128*128 = 524288
  const int ntok = in_sizes[0] / D_DIM;        // 8192

  htl_abs_partial<<<256, 256, 0, stream>>>(w, part, nW);
  htl_abs_final  <<<1,   256, 0, stream>>>(part, scalep, 1.0f / (float)nW);
  htl_quantize   <<<(nW + 255) / 256, 256, 0, stream>>>(w, scalep, wq, nW);
  htl_fused      <<<ntok / MTILE, 256, 0, stream>>>(x, wq, alpha, beta, scalep, out);
}
